// GNNEncoder_29197187678590
// MI455X (gfx1250) — matrix operand order repaired and checked by reading, not compiled, not run on hardware
//
#include <hip/hip_runtime.h>
#include <hip/hip_bf16.h>
#include <cstdint>

#define V_NODES 10000
#define E_EDGES 160000
#define H 256
#define LN_EPS 1e-5f

typedef __attribute__((ext_vector_type(16))) __bf16 v16bf;
typedef __attribute__((ext_vector_type(2)))  __bf16 v2bf;
typedef __attribute__((ext_vector_type(8)))  float  v8f;

__device__ __forceinline__ unsigned short f2bf(float f) {
    union { float f; unsigned u; } x; x.f = f;
    unsigned u = x.u;
    // round-to-nearest-even bf16 truncation
    unsigned r = (u + 0x7FFFu + ((u >> 16) & 1u)) >> 16;
    return (unsigned short)r;
}

__device__ __forceinline__ unsigned pk2(float a, float b) {
#if __has_builtin(__builtin_amdgcn_cvt_pk_bf16_f32)
    v2bf t = __builtin_amdgcn_cvt_pk_bf16_f32(a, b);
    return __builtin_bit_cast(unsigned, t);
#else
    return (unsigned)f2bf(a) | ((unsigned)f2bf(b) << 16);
#endif
}

// Truncate a generic pointer to a shared-memory object down to its 32-bit LDS
// offset (flat LDS addresses carry the offset in addr[31:0]).
__device__ __forceinline__ unsigned lds_off(const void* p) {
    return (unsigned)(size_t)p;
}

// CDNA5 async copy: global -> LDS, 16 bytes per lane, tracked by ASYNCcnt.
__device__ __forceinline__ void async_copy_b128(unsigned ldsByteOff, const void* gaddr) {
    asm volatile("global_load_async_to_lds_b128 %0, %1, off"
                 :: "v"(ldsByteOff), "v"((unsigned long long)(size_t)gaddr)
                 : "memory");
}

__device__ __forceinline__ void wait_async0() {
    asm volatile("s_wait_asynccnt 0x0" ::: "memory");
}

// ---------------------------------------------------------------------------
// prep: zero agg, transpose-convert the 5 weight matrices to bf16 Wt[k][n]
// ---------------------------------------------------------------------------
__global__ void prep_kernel(const float* __restrict__ W0, const float* __restrict__ W1,
                            const float* __restrict__ W2, const float* __restrict__ W3,
                            const float* __restrict__ W4,
                            unsigned short* __restrict__ Wt, float* __restrict__ agg) {
    int idx = blockIdx.x * blockDim.x + threadIdx.x;
    if (idx < V_NODES * H) agg[idx] = 0.0f;
    const int total = 5 * H * H;
    if (idx < total) {
        int w = idx / (H * H);
        int r = idx % (H * H);
        int n = r / H;
        int k = r % H;
        const float* W = (w == 0) ? W0 : (w == 1) ? W1 : (w == 2) ? W2 : (w == 3) ? W3 : W4;
        Wt[(size_t)w * H * H + (size_t)k * H + n] = f2bf(W[(size_t)n * H + k]);
    }
}

// ---------------------------------------------------------------------------
// Tiled bf16 WMMA GEMM:  Y[m][n] = sum_k X[m][k] * W[n][k] + bias[n]
//   X:  M x H fp32 (async-copied to LDS, converted to bf16 in registers)
//   Wt: H x H bf16, pre-transposed -> Wt[k][n]; 128-col panel staged in LDS once
// Edge mode (srcIdx != nullptr): Y += Ah[dst[m]][n] + Bh[src[m]][n]
// Block: 256 threads (8 waves). Block tile 128(M) x 128(N); wave tile 16 x 128.
// LDS: 2 x 16 KB X double-buffer + 64 KB weight panel = 96 KB.
// ---------------------------------------------------------------------------
__global__ void gemm_kernel(const float* __restrict__ X, int M,
                            const unsigned short* __restrict__ Wt,
                            const float* __restrict__ bias,
                            float* __restrict__ Y,
                            const int* __restrict__ srcIdx,
                            const int* __restrict__ dstIdx,
                            const float* __restrict__ Ah,
                            const float* __restrict__ Bh) {
    __shared__ __align__(16) float          Xs[2][128 * 32];   // 2 x 16 KB fp32 tiles
    __shared__ __align__(16) unsigned short Wsh[H * 128];      // 64 KB bf16 weight panel

    const int tid  = threadIdx.x;
    const int lane = tid & 31;
    const int wave = tid >> 5;
    const int row0  = blockIdx.x * 128;
    const int ncol0 = blockIdx.y * 128;

    // ---- prologue: async-stage the whole 256x128 weight panel into LDS ----
    #pragma unroll
    for (int i = 0; i < 16; ++i) {
        int id = tid + i * 256;            // 0..4095 chunks of 8 bf16
        int r  = id >> 4;                  // k row 0..255
        int sg = id & 15;                  // 8-bf16 segment 0..15
        const unsigned short* g = Wt + (size_t)r * H + ncol0 + sg * 8;
        async_copy_b128(lds_off(&Wsh[r * 128 + sg * 8]), g);
    }

    // ---- async stage of one 128x32 fp32 X tile ----
    auto issue_X = [&](int kstep, int b) {
        const int k0 = kstep * 32;
        #pragma unroll
        for (int i = 0; i < 4; ++i) {
            int id = tid + i * 256;        // 0..1023 chunks of 4 floats
            int r  = id >> 3;              // tile row 0..127
            int sg = id & 7;               // float4 segment 0..7
            int grow = row0 + r;
            if (grow >= M) grow = M - 1;
            const float* g = X + (size_t)grow * H + k0 + sg * 4;
            async_copy_b128(lds_off(&Xs[b][r * 32 + sg * 4]), g);
        }
    };
    issue_X(0, 0);

    v8f acc[8] = {};
    union Frag { unsigned u[8]; uint4 q[2]; v16bf v; };

    const int mrow = wave * 16 + (lane & 15);   // A: lane -> M row of wave tile
    const int half = lane >> 4;                 // A: half-lane K split

    #pragma unroll 2
    for (int kstep = 0; kstep < H / 32; ++kstep) {
        wait_async0();        // this wave's async copies have landed
        __syncthreads();      // everyone's have landed; prior reads done

        if (kstep + 1 < H / 32) issue_X(kstep + 1, (kstep + 1) & 1);

        // ---- A fragment: 16 fp32 from LDS -> 16 bf16 packed in 8 regs ----
        const float4* Xf4 = (const float4*)Xs[kstep & 1];
        const int abase = mrow * 8 + half * 2;       // float4 index
        float4 c0 = Xf4[abase + 0];                  // K = half*8 .. +3
        float4 c1 = Xf4[abase + 1];                  // K = half*8+4 .. +7
        float4 c2 = Xf4[abase + 4];                  // K = 16+half*8 ..
        float4 c3 = Xf4[abase + 5];
        Frag fa;
        fa.u[0] = pk2(c0.x, c0.y); fa.u[1] = pk2(c0.z, c0.w);
        fa.u[2] = pk2(c1.x, c1.y); fa.u[3] = pk2(c1.z, c1.w);
        fa.u[4] = pk2(c2.x, c2.y); fa.u[5] = pk2(c2.z, c2.w);
        fa.u[6] = pk2(c3.x, c3.y); fa.u[7] = pk2(c3.z, c3.w);

        // ---- B fragments: lane -> N column, elements over K (same K order as A) ----
        const int kbase = kstep * 32 + half * 8;
        const int bn = lane & 15;
        #pragma unroll
        for (int s = 0; s < 8; ++s) {
            Frag fb;
            const int col = s * 16 + bn;
            #pragma unroll
            for (int j = 0; j < 8; ++j) {
                const int k = kbase + (j < 4 ? 2 * j : 16 + 2 * (j - 4));
                fb.u[j] = (unsigned)Wsh[k * 128 + col] | ((unsigned)Wsh[(k + 1) * 128 + col] << 16);
            }
            acc[s] = __builtin_amdgcn_wmma_f32_16x16x32_bf16(
                false, fa.v, false, fb.v, (short)0, acc[s], false, false);
        }
    }

    // ---- epilogue: D layout VGPR r -> M = r + 8*(lane>=16), N = lane&15 ----
    const int nl = lane & 15;
    const int mh = (lane >> 4) * 8;
    float bvals[8];
    #pragma unroll
    for (int s = 0; s < 8; ++s) bvals[s] = bias[ncol0 + s * 16 + nl];

    #pragma unroll
    for (int r = 0; r < 8; ++r) {
        const int row = row0 + wave * 16 + mh + r;
        if (row < M) {
            size_t dOff = 0, sOff = 0;
            const bool gath = (srcIdx != nullptr);
            if (gath) {
                dOff = (size_t)dstIdx[row] * H;
                sOff = (size_t)srcIdx[row] * H;
            }
            #pragma unroll
            for (int s = 0; s < 8; ++s) {
                const int col = ncol0 + s * 16 + nl;
                float v = acc[s][r] + bvals[s];
                if (gath) v += Ah[dOff + col] + Bh[sOff + col];
                Y[(size_t)row * H + col] = v;
            }
        }
    }
}

// ---------------------------------------------------------------------------
// Per-edge: gate+scatter into agg, LayerNorm+ReLU+residual -> e_out (in place)
// ---------------------------------------------------------------------------
__global__ void edge_kernel(float* __restrict__ enew_io,      // e_new in, e_out out
                            const float* __restrict__ e_in,
                            const float* __restrict__ Vh,
                            const int* __restrict__ eidx,     // (2,E) row-major
                            float* __restrict__ agg,
                            const float* __restrict__ ge,
                            const float* __restrict__ be) {
    __shared__ float red[256];
    const int e = blockIdx.x;
    const int j = threadIdx.x;
    const float x = enew_io[(size_t)e * H + j];
    const int s = eidx[e];            // aggregation target
    const int d = eidx[E_EDGES + e];  // neighbor source

    const float g = 1.0f / (1.0f + __expf(-x));
    atomicAdd(&agg[(size_t)s * H + j], g * Vh[(size_t)d * H + j]);

    red[j] = x;
    __syncthreads();
    for (int off = 128; off > 0; off >>= 1) {
        if (j < off) red[j] += red[j + off];
        __syncthreads();
    }
    const float mu = red[0] * (1.0f / H);
    __syncthreads();
    const float dv = x - mu;
    red[j] = dv * dv;
    __syncthreads();
    for (int off = 128; off > 0; off >>= 1) {
        if (j < off) red[j] += red[j + off];
        __syncthreads();
    }
    const float var = red[0] * (1.0f / H);
    const float y = dv * rsqrtf(var + LN_EPS) * ge[j] + be[j];
    enew_io[(size_t)e * H + j] = e_in[(size_t)e * H + j] + fmaxf(y, 0.0f);
}

// ---------------------------------------------------------------------------
// Per-node: x = Uh + agg; LayerNorm+ReLU+residual -> h_out
// ---------------------------------------------------------------------------
__global__ void node_kernel(const float* __restrict__ Uh,
                            const float* __restrict__ agg,
                            const float* __restrict__ h_in,
                            float* __restrict__ h_out,
                            const float* __restrict__ gh,
                            const float* __restrict__ bh) {
    __shared__ float red[256];
    const int v = blockIdx.x;
    const int j = threadIdx.x;
    const float x = Uh[(size_t)v * H + j] + agg[(size_t)v * H + j];

    red[j] = x;
    __syncthreads();
    for (int off = 128; off > 0; off >>= 1) {
        if (j < off) red[j] += red[j + off];
        __syncthreads();
    }
    const float mu = red[0] * (1.0f / H);
    __syncthreads();
    const float dv = x - mu;
    red[j] = dv * dv;
    __syncthreads();
    for (int off = 128; off > 0; off >>= 1) {
        if (j < off) red[j] += red[j + off];
        __syncthreads();
    }
    const float var = red[0] * (1.0f / H);
    const float y = dv * rsqrtf(var + LN_EPS) * gh[j] + bh[j];
    h_out[(size_t)v * H + j] = h_in[(size_t)v * H + j] + fmaxf(y, 0.0f);
}

// ---------------------------------------------------------------------------
extern "C" void kernel_launch(void* const* d_in, const int* in_sizes, int n_in,
                              void* d_out, int out_size, void* d_ws, size_t ws_size,
                              hipStream_t stream) {
    const float* h    = (const float*)d_in[0];
    const float* e    = (const float*)d_in[1];
    const int*   eidx = (const int*)d_in[2];
    const float* Wu = (const float*)d_in[3];  const float* bu = (const float*)d_in[4];
    const float* Wv = (const float*)d_in[5];  const float* bv = (const float*)d_in[6];
    const float* Wa = (const float*)d_in[7];  const float* ba = (const float*)d_in[8];
    const float* Wb = (const float*)d_in[9];  const float* bb = (const float*)d_in[10];
    const float* Wc = (const float*)d_in[11]; const float* bc = (const float*)d_in[12];
    const float* gh = (const float*)d_in[13]; const float* bh = (const float*)d_in[14];
    const float* ge = (const float*)d_in[15]; const float* be = (const float*)d_in[16];

    float* h_out = (float*)d_out;                           // V*H
    float* e_new = (float*)d_out + (size_t)V_NODES * H;     // E*H (e_new -> e_out in place)

    const size_t VH = (size_t)V_NODES * H;
    char* ws = (char*)d_ws;
    unsigned short* Wt = (unsigned short*)ws;  ws += (size_t)5 * H * H * sizeof(unsigned short);
    float* Uh  = (float*)ws; ws += VH * sizeof(float);
    float* Vh  = (float*)ws; ws += VH * sizeof(float);
    float* Ah  = (float*)ws; ws += VH * sizeof(float);
    float* Bh  = (float*)ws; ws += VH * sizeof(float);
    float* agg = (float*)ws; ws += VH * sizeof(float);

    // prep: zero agg + weight transpose/convert (needs max(V*H, 5*H*H) threads)
    prep_kernel<<<(V_NODES * H + 255) / 256, 256, 0, stream>>>(Wu, Wv, Wa, Wb, Wc, Wt, agg);

    // node GEMMs: Uh, Vh, Ah, Bh
    dim3 gn((V_NODES + 127) / 128, H / 128);
    gemm_kernel<<<gn, 256, 0, stream>>>(h, V_NODES, Wt + 0 * H * H, bu, Uh,
                                        nullptr, nullptr, nullptr, nullptr);
    gemm_kernel<<<gn, 256, 0, stream>>>(h, V_NODES, Wt + 1 * H * H, bv, Vh,
                                        nullptr, nullptr, nullptr, nullptr);
    gemm_kernel<<<gn, 256, 0, stream>>>(h, V_NODES, Wt + 2 * H * H, ba, Ah,
                                        nullptr, nullptr, nullptr, nullptr);
    gemm_kernel<<<gn, 256, 0, stream>>>(h, V_NODES, Wt + 3 * H * H, bb, Bh,
                                        nullptr, nullptr, nullptr, nullptr);

    // edge GEMM, fused gather: e_new = e*Wc^T + bc + Ah[dst] + Bh[src]
    dim3 gE(E_EDGES / 128, H / 128);
    gemm_kernel<<<gE, 256, 0, stream>>>(e, E_EDGES, Wt + (size_t)4 * H * H, bc, e_new,
                                        eidx, eidx + E_EDGES, Ah, Bh);

    // edge pass: gate + scatter + LN/ReLU/residual (e_out in place)
    edge_kernel<<<E_EDGES, 256, 0, stream>>>(e_new, e, Vh, eidx, agg, ge, be);

    // node pass: LN/ReLU/residual -> h_out
    node_kernel<<<V_NODES, 256, 0, stream>>>(Uh, agg, h, h_out, gh, bh);
}
